// BitNetAttention_1340029796977
// MI455X (gfx1250) — compile-verified, matrix operand order checked by reading, not run on hardware
//
#include <hip/hip_runtime.h>
#include <hip/hip_bf16.h>
#include <cstdint>

constexpr int BATCH = 2;
constexpr int SEQ   = 2048;
constexpr int HDIM  = 2048;
constexpr int NH    = 16;
constexpr int NKV   = 8;
constexpr int HD    = 128;
constexpr int TOK   = BATCH * SEQ;

typedef __attribute__((ext_vector_type(8)))  int      v8i;
typedef __attribute__((ext_vector_type(8)))  float    v8f;
typedef __attribute__((ext_vector_type(16))) _Float16 v16h;
typedef __attribute__((ext_vector_type(8)))  _Float16 v8h;

__device__ inline v16h load_frag16(const _Float16* p, int half) {
  union { v16h v; v8h h[2]; } u;
  u.h[0] = *reinterpret_cast<const v8h*>(p + half * 8);
  u.h[1] = *reinterpret_cast<const v8h*>(p + 16 + half * 8);
  return u.v;
}

__device__ inline v8i load_frag8(const int8_t* row, int k0, int half) {
  v8i a;
#pragma unroll
  for (int c = 0; c < 4; ++c) {
    int2 d = *reinterpret_cast<const int2*>(row + k0 + c * 16 + half * 8);
    a[2 * c]     = d.x;
    a[2 * c + 1] = d.y;
  }
  return a;
}

__device__ inline void async_copy_b128(uint32_t lds_byte_addr, const void* gptr) {
  asm volatile("global_load_async_to_lds_b128 %0, %1, off"
               :: "v"(lds_byte_addr), "v"((uint64_t)(uintptr_t)gptr)
               : "memory");
}

__global__ void bn_init_acc(float* acc, int n) {
  int i = blockIdx.x * blockDim.x + threadIdx.x;
  if (i < n) acc[i] = 0.0f;
}

__global__ __launch_bounds__(256) void bn_wstats(const float* __restrict__ w, int n,
                                                 float* __restrict__ acc2) {
  __shared__ float rs[256], ra[256];
  float s = 0.f, sa = 0.f;
  for (int i = blockIdx.x * blockDim.x + threadIdx.x; i < n; i += gridDim.x * blockDim.x) {
    float v = w[i];
    s += v; sa += fabsf(v);
  }
  rs[threadIdx.x] = s; ra[threadIdx.x] = sa;
  __syncthreads();
  for (int off = 128; off; off >>= 1) {
    if (threadIdx.x < off) {
      rs[threadIdx.x] += rs[threadIdx.x + off];
      ra[threadIdx.x] += ra[threadIdx.x + off];
    }
    __syncthreads();
  }
  if (threadIdx.x == 0) {
    atomicAdd(&acc2[0], rs[0]);
    atomicAdd(&acc2[1], ra[0]);
  }
}

__global__ __launch_bounds__(256) void bn_wpack(const float* __restrict__ w,
                                                const float* __restrict__ acc2,
                                                float inv_cnt, int8_t* __restrict__ w8, int n) {
  float e = acc2[0] * inv_cnt;
  for (int i = blockIdx.x * blockDim.x + threadIdx.x; i < n; i += gridDim.x * blockDim.x)
    w8[i] = (w[i] >= e) ? (int8_t)1 : (int8_t)-1;
}

__global__ __launch_bounds__(256) void bn_actq3(const float* __restrict__ x,
                                                const float* __restrict__ n0,
                                                const float* __restrict__ n1,
                                                const float* __restrict__ n2,
                                                int8_t* __restrict__ q0, int8_t* __restrict__ q1,
                                                int8_t* __restrict__ q2,
                                                float* __restrict__ i0, float* __restrict__ i1,
                                                float* __restrict__ i2) {
  const int row = blockIdx.x, t = threadIdx.x;
  __shared__ float red[256];
  __shared__ float bc;
  float xr[8], ss = 0.f;
#pragma unroll
  for (int i = 0; i < 8; ++i) {
    int col = t + i * 256;
    float v = x[(size_t)row * HDIM + col];
    v = fminf(fmaxf(v, -100.f), 100.f);
    xr[i] = v; ss += v * v;
  }
  red[t] = ss; __syncthreads();
  for (int off = 128; off; off >>= 1) { if (t < off) red[t] += red[t + off]; __syncthreads(); }
  if (t == 0) bc = rsqrtf(fmaxf(red[0] * (1.0f / HDIM), 1e-5f) + 1e-5f);
  __syncthreads();
  const float rs = bc;

  const float* nw[3] = {n0, n1, n2};
  int8_t* qs[3] = {q0, q1, q2};
  float* iv[3] = {i0, i1, i2};
  for (int w = 0; w < 3; ++w) {
    float y[8], mx = 0.f;
#pragma unroll
    for (int i = 0; i < 8; ++i) {
      int col = t + i * 256;
      float v = fminf(fmaxf(xr[i] * rs, -10.f), 10.f) * nw[w][col];
      v = fminf(fmaxf(v, -50.f), 50.f);
      y[i] = v; mx = fmaxf(mx, fabsf(v));
    }
    __syncthreads();
    red[t] = mx; __syncthreads();
    for (int off = 128; off; off >>= 1) { if (t < off) red[t] = fmaxf(red[t], red[t + off]); __syncthreads(); }
    if (t == 0) bc = fmaxf(red[0], 1e-4f);
    __syncthreads();
    const float m = bc, sc = 127.f / m;
#pragma unroll
    for (int i = 0; i < 8; ++i) {
      int col = t + i * 256;
      qs[w][(size_t)row * HDIM + col] = (int8_t)fminf(fmaxf(rintf(y[i] * sc), -128.f), 127.f);
    }
    if (t == 0) iv[w][row] = m * (1.0f / 127.0f);
    __syncthreads();
  }
}

__global__ __launch_bounds__(256) void bn_actq1(const float* __restrict__ x,
                                                const float* __restrict__ nw,
                                                int8_t* __restrict__ q, float* __restrict__ iv) {
  const int row = blockIdx.x, t = threadIdx.x;
  __shared__ float red[256];
  __shared__ float bc;
  float xr[8], ss = 0.f;
#pragma unroll
  for (int i = 0; i < 8; ++i) {
    int col = t + i * 256;
    float v = x[(size_t)row * HDIM + col];
    v = fminf(fmaxf(v, -100.f), 100.f);
    xr[i] = v; ss += v * v;
  }
  red[t] = ss; __syncthreads();
  for (int off = 128; off; off >>= 1) { if (t < off) red[t] += red[t + off]; __syncthreads(); }
  if (t == 0) bc = rsqrtf(fmaxf(red[0] * (1.0f / HDIM), 1e-5f) + 1e-5f);
  __syncthreads();
  const float rs = bc;
  float y[8], mx = 0.f;
#pragma unroll
  for (int i = 0; i < 8; ++i) {
    int col = t + i * 256;
    float v = fminf(fmaxf(xr[i] * rs, -10.f), 10.f) * nw[col];
    v = fminf(fmaxf(v, -50.f), 50.f);
    y[i] = v; mx = fmaxf(mx, fabsf(v));
  }
  __syncthreads();
  red[t] = mx; __syncthreads();
  for (int off = 128; off; off >>= 1) { if (t < off) red[t] = fmaxf(red[t], red[t + off]); __syncthreads(); }
  if (t == 0) bc = fmaxf(red[0], 1e-4f);
  __syncthreads();
  const float m = bc, sc = 127.f / m;
#pragma unroll
  for (int i = 0; i < 8; ++i) {
    int col = t + i * 256;
    q[(size_t)row * HDIM + col] = (int8_t)fminf(fmaxf(rintf(y[i] * sc), -128.f), 127.f);
  }
  if (t == 0) iv[row] = m * (1.0f / 127.0f);
}

__global__ __launch_bounds__(256) void bn_gemm_iu8(const int8_t* __restrict__ A,
                                                   const int8_t* __restrict__ W,
                                                   const float* __restrict__ ainv,
                                                   const float* __restrict__ wacc2, float inv_cnt,
                                                   float* __restrict__ C, int M, int N, int K) {
  const int lane = threadIdx.x & 31, wv = threadIdx.x >> 5;
  const int tile = blockIdx.x * (blockDim.x >> 5) + wv;
  const int ntn4 = N >> 6;
  const int mt = tile / ntn4, nt4 = tile % ntn4;
  if (mt * 16 >= M) return;
  const int half = lane >> 4, ln = lane & 15;

  const int8_t* arow = A + (size_t)(mt * 16 + ln) * K;
  const int8_t* wr0 = W + (size_t)(nt4 * 64 + 0 * 16 + ln) * K;
  const int8_t* wr1 = W + (size_t)(nt4 * 64 + 1 * 16 + ln) * K;
  const int8_t* wr2 = W + (size_t)(nt4 * 64 + 2 * 16 + ln) * K;
  const int8_t* wr3 = W + (size_t)(nt4 * 64 + 3 * 16 + ln) * K;

  v8i acc0 = {0, 0, 0, 0, 0, 0, 0, 0};
  v8i acc1 = acc0, acc2 = acc0, acc3 = acc0;
  for (int k0 = 0; k0 < K; k0 += 64) {
    v8i af = load_frag8(arow, k0, half);
    v8i b0 = load_frag8(wr0, k0, half);
    v8i b1 = load_frag8(wr1, k0, half);
    v8i b2 = load_frag8(wr2, k0, half);
    v8i b3 = load_frag8(wr3, k0, half);
    acc0 = __builtin_amdgcn_wmma_i32_16x16x64_iu8(true, af, true, b0, acc0, false, false);
    acc1 = __builtin_amdgcn_wmma_i32_16x16x64_iu8(true, af, true, b1, acc1, false, false);
    acc2 = __builtin_amdgcn_wmma_i32_16x16x64_iu8(true, af, true, b2, acc2, false, false);
    acc3 = __builtin_amdgcn_wmma_i32_16x16x64_iu8(true, af, true, b3, acc3, false, false);
  }
  const float wscale = fmaxf(wacc2[1] * inv_cnt, 1e-8f);
  v8i accs[4] = {acc0, acc1, acc2, acc3};
#pragma unroll
  for (int j = 0; j < 4; ++j)
#pragma unroll
    for (int r = 0; r < 8; ++r) {
      int row = mt * 16 + r + half * 8;
      int col = nt4 * 64 + j * 16 + ln;
      C[(size_t)row * N + col] = (float)accs[j][r] * ainv[row] * wscale;
    }
}

__global__ __launch_bounds__(256) void bn_rope(const float* __restrict__ src,
                                               _Float16* __restrict__ dst, int nheads, int total) {
  int idx = blockIdx.x * blockDim.x + threadIdx.x;
  if (idx >= total) return;
  int d = idx & 127;
  int rem = idx >> 7;
  int h = rem % nheads;
  int rem2 = rem / nheads;
  int s = rem2 % SEQ;
  int b = rem2 / SEQ;
  float x = src[idx];
  int d2 = (d < 64) ? d + 64 : d - 64;
  float x2 = src[idx - d + d2];
  float inv = __expf(-(float)(d & 63) * 0.14391156f);
  float ang = (float)s * inv;
  float sn, cs;
  __sincosf(ang, &sn, &cs);
  float r = x * cs + ((d < 64) ? -x2 : x2) * sn;
  dst[(((size_t)(b * nheads + h)) * SEQ + s) * HD + d] = (_Float16)r;
}

__global__ __launch_bounds__(256) void bn_vt(const float* __restrict__ src,
                                             _Float16* __restrict__ dst, int total) {
  int idx = blockIdx.x * blockDim.x + threadIdx.x;
  if (idx >= total) return;
  int d = idx & 127;
  int rem = idx >> 7;
  int h = rem % NKV;
  int rem2 = rem / NKV;
  int s = rem2 % SEQ;
  int b = rem2 / SEQ;
  dst[(((size_t)(b * NKV + h)) * HD + d) * SEQ + s] = (_Float16)src[idx];
}

__global__ __launch_bounds__(128) void bn_flash(const _Float16* __restrict__ qh,
                                                const _Float16* __restrict__ kh,
                                                const _Float16* __restrict__ vth,
                                                const float* __restrict__ mask,
                                                float* __restrict__ out) {
  const int tid = threadIdx.x;
  const int lane = tid & 31, wv = tid >> 5;
  const int half = lane >> 4, ln = lane & 15;
  const int nqb = SEQ / 64;
  const int qb = blockIdx.x % nqb;
  const int bh = blockIdx.x / nqb;
  const int h = bh % NH, b = bh / NH;
  const int hkv = h >> 1;
  const int q0 = qb * 64 + wv * 16;

  __shared__ _Float16 Ksh[32 * 128];
  __shared__ _Float16 Vsh[128 * 32];
  __shared__ _Float16 ldsP[4][16 * 32];
  _Float16* P = ldsP[wv];

  const _Float16* Qb = qh + ((size_t)(b * NH + h)) * SEQ * HD;
  const _Float16* Kb = kh + ((size_t)(b * NKV + hkv)) * SEQ * HD;
  const _Float16* Vb = vth + ((size_t)(b * NKV + hkv)) * HD * SEQ;

  const uint32_t ldsK = (uint32_t)(uintptr_t)(void*)Ksh;
  const uint32_t ldsV = (uint32_t)(uintptr_t)(void*)Vsh;

  v16h Aq[4];
#pragma unroll
  for (int kc = 0; kc < 4; ++kc)
    Aq[kc] = load_frag16(Qb + (size_t)(q0 + ln) * HD + kc * 32, half);

  float rm[8], rl[8];
  v8f acc[8];
#pragma unroll
  for (int r = 0; r < 8; ++r) { rm[r] = -1e30f; rl[r] = 0.f; }
#pragma unroll
  for (int t = 0; t < 8; ++t) acc[t] = (v8f){0, 0, 0, 0, 0, 0, 0, 0};

  const float sm_scale = 0.08838834764831845f;

  for (int kb = 0; kb < SEQ; kb += 32) {
#pragma unroll
    for (int c = 0; c < 4; ++c) {
      int lid = tid + c * 128;
      const char* gk = (const char*)(Kb + (size_t)(kb + (lid >> 4)) * HD) + (lid & 15) * 16;
      async_copy_b128(ldsK + (uint32_t)lid * 16u, gk);
      int rowv = lid >> 2, seg = lid & 3;
      const char* gv = (const char*)(Vb + (size_t)rowv * SEQ + kb) + seg * 16;
      async_copy_b128(ldsV + (uint32_t)(rowv * 64 + seg * 16), gv);
    }
    asm volatile("s_wait_asynccnt 0" ::: "memory");
    __syncthreads();

    float xs[2][8];
#pragma unroll
    for (int j = 0; j < 2; ++j) {
      v8f s = (v8f){0, 0, 0, 0, 0, 0, 0, 0};
#pragma unroll
      for (int kc = 0; kc < 4; ++kc) {
        v16h Bk = load_frag16(Ksh + (j * 16 + ln) * HD + kc * 32, half);
        s = __builtin_amdgcn_wmma_f32_16x16x32_f16(false, Aq[kc], false, Bk, (short)0, s,
                                                   false, false);
      }
#pragma unroll
      for (int r = 0; r < 8; ++r) {
        int qrow = q0 + r + half * 8;
        xs[j][r] = s[r] * sm_scale + mask[(size_t)qrow * SEQ + kb + j * 16 + ln];
      }
    }
#pragma unroll
    for (int r = 0; r < 8; ++r) {
      float v = fmaxf(xs[0][r], xs[1][r]);
      v = fmaxf(v, __shfl_xor(v, 1, 16));
      v = fmaxf(v, __shfl_xor(v, 2, 16));
      v = fmaxf(v, __shfl_xor(v, 4, 16));
      v = fmaxf(v, __shfl_xor(v, 8, 16));
      float mn = fmaxf(rm[r], v);
      float corr = __expf(rm[r] - mn);
      rm[r] = mn;
      float p0 = __expf(xs[0][r] - mn);
      float p1 = __expf(xs[1][r] - mn);
      xs[0][r] = p0; xs[1][r] = p1;
      float ps = p0 + p1;
      ps += __shfl_xor(ps, 1, 16);
      ps += __shfl_xor(ps, 2, 16);
      ps += __shfl_xor(ps, 4, 16);
      ps += __shfl_xor(ps, 8, 16);
      rl[r] = rl[r] * corr + ps;
#pragma unroll
      for (int t = 0; t < 8; ++t) acc[t][r] *= corr;
    }
#pragma unroll
    for (int j = 0; j < 2; ++j)
#pragma unroll
      for (int r = 0; r < 8; ++r)
        P[(r + half * 8) * 32 + j * 16 + ln] = (_Float16)xs[j][r];
    __asm__ volatile("s_wait_dscnt 0" ::: "memory");
    v16h Pf = load_frag16(P + ln * 32, half);
#pragma unroll
    for (int t = 0; t < 8; ++t) {
      v16h Bv = load_frag16(Vsh + (t * 16 + ln) * 32, half);
      acc[t] = __builtin_amdgcn_wmma_f32_16x16x32_f16(false, Pf, false, Bv, (short)0, acc[t],
                                                      false, false);
    }
    __syncthreads();
  }
#pragma unroll
  for (int r = 0; r < 8; ++r) {
    float inv = 1.0f / rl[r];
    int qrow = q0 + r + half * 8;
    float* orow = out + ((size_t)(b * SEQ + qrow)) * HDIM + h * HD;
#pragma unroll
    for (int t = 0; t < 8; ++t) orow[t * 16 + ln] = acc[t][r] * inv;
  }
}

extern "C" void kernel_launch(void* const* d_in, const int* in_sizes, int n_in,
                              void* d_out, int out_size, void* d_ws, size_t ws_size,
                              hipStream_t stream) {
  const float* x    = (const float*)d_in[0];
  const float* mask = (const float*)d_in[1];
  const float* wq   = (const float*)d_in[2];
  const float* wk   = (const float*)d_in[3];
  const float* wv   = (const float*)d_in[4];
  const float* wo   = (const float*)d_in[5];
  const float* nq   = (const float*)d_in[6];
  const float* nk   = (const float*)d_in[7];
  const float* nv   = (const float*)d_in[8];
  const float* no   = (const float*)d_in[9];
  float* out = (float*)d_out;

  char* ws = (char*)d_ws;
  size_t off = 0;
  auto alloc = [&](size_t bytes) { size_t o = off; off = (off + bytes + 255) & ~(size_t)255; return o; };

  const size_t o_acc  = alloc(8 * sizeof(float));
  const size_t o_xq0  = alloc((size_t)TOK * HDIM);
  const size_t o_xq1  = alloc((size_t)TOK * HDIM);
  const size_t o_xq2  = alloc((size_t)TOK * HDIM);
  const size_t o_xi0  = alloc((size_t)TOK * 4);
  const size_t o_xi1  = alloc((size_t)TOK * 4);
  const size_t o_xi2  = alloc((size_t)TOK * 4);
  const size_t o_w8q  = alloc((size_t)HDIM * HDIM);
  const size_t o_w8k  = alloc((size_t)NKV * HD * HDIM);
  const size_t o_w8v  = alloc((size_t)NKV * HD * HDIM);
  const size_t o_w8o  = alloc((size_t)HDIM * HDIM);
  const size_t o_qf   = alloc((size_t)TOK * HDIM * 4);
  const size_t o_kf   = alloc((size_t)TOK * NKV * HD * 4);
  const size_t o_vf   = alloc((size_t)TOK * NKV * HD * 4);
  const size_t o_qh   = alloc((size_t)TOK * HDIM * 2);
  const size_t o_kh   = alloc((size_t)TOK * NKV * HD * 2);
  const size_t o_vth  = alloc((size_t)TOK * NKV * HD * 2);
  (void)ws_size; (void)in_sizes; (void)n_in; (void)out_size;

  float*  acc  = (float*)(ws + o_acc);
  int8_t* xq0  = (int8_t*)(ws + o_xq0);
  int8_t* xq1  = (int8_t*)(ws + o_xq1);
  int8_t* xq2  = (int8_t*)(ws + o_xq2);
  float*  xi0  = (float*)(ws + o_xi0);
  float*  xi1  = (float*)(ws + o_xi1);
  float*  xi2  = (float*)(ws + o_xi2);
  int8_t* w8q  = (int8_t*)(ws + o_w8q);
  int8_t* w8k  = (int8_t*)(ws + o_w8k);
  int8_t* w8v  = (int8_t*)(ws + o_w8v);
  int8_t* w8o  = (int8_t*)(ws + o_w8o);
  float*  qf   = (float*)(ws + o_qf);
  float*  kf   = (float*)(ws + o_kf);
  float*  vf   = (float*)(ws + o_vf);
  _Float16* qh  = (_Float16*)(ws + o_qh);
  _Float16* kh  = (_Float16*)(ws + o_kh);
  _Float16* vth = (_Float16*)(ws + o_vth);
  float*  attn_out = qf;
  int8_t* ao8      = (int8_t*)kf;
  float*  aoinv    = vf;

  const int nQO = HDIM * HDIM;
  const int nKV = NKV * HD * HDIM;

  hipLaunchKernelGGL(bn_init_acc, dim3(1), dim3(32), 0, stream, acc, 8);
  hipLaunchKernelGGL(bn_wstats, dim3(512), dim3(256), 0, stream, wq, nQO, acc + 0);
  hipLaunchKernelGGL(bn_wstats, dim3(512), dim3(256), 0, stream, wk, nKV, acc + 2);
  hipLaunchKernelGGL(bn_wstats, dim3(512), dim3(256), 0, stream, wv, nKV, acc + 4);
  hipLaunchKernelGGL(bn_wstats, dim3(512), dim3(256), 0, stream, wo, nQO, acc + 6);

  hipLaunchKernelGGL(bn_wpack, dim3(2048), dim3(256), 0, stream, wq, acc + 0, 1.0f / nQO, w8q, nQO);
  hipLaunchKernelGGL(bn_wpack, dim3(2048), dim3(256), 0, stream, wk, acc + 2, 1.0f / nKV, w8k, nKV);
  hipLaunchKernelGGL(bn_wpack, dim3(2048), dim3(256), 0, stream, wv, acc + 4, 1.0f / nKV, w8v, nKV);
  hipLaunchKernelGGL(bn_wpack, dim3(2048), dim3(256), 0, stream, wo, acc + 6, 1.0f / nQO, w8o, nQO);

  hipLaunchKernelGGL(bn_actq3, dim3(TOK), dim3(256), 0, stream, x, nq, nk, nv,
                     xq0, xq1, xq2, xi0, xi1, xi2);

  {
    int tilesQ = (TOK / 16) * (HDIM / 64);
    hipLaunchKernelGGL(bn_gemm_iu8, dim3(tilesQ / 8), dim3(256), 0, stream,
                       xq0, w8q, xi0, acc + 0, 1.0f / nQO, qf, TOK, HDIM, HDIM);
    int tilesK = (TOK / 16) * ((NKV * HD) / 64);
    hipLaunchKernelGGL(bn_gemm_iu8, dim3(tilesK / 8), dim3(256), 0, stream,
                       xq1, w8k, xi1, acc + 2, 1.0f / nKV, kf, TOK, NKV * HD, HDIM);
    hipLaunchKernelGGL(bn_gemm_iu8, dim3(tilesK / 8), dim3(256), 0, stream,
                       xq2, w8v, xi2, acc + 4, 1.0f / nKV, vf, TOK, NKV * HD, HDIM);
  }

  {
    int totQ = TOK * HDIM;
    int totK = TOK * NKV * HD;
    hipLaunchKernelGGL(bn_rope, dim3(totQ / 256), dim3(256), 0, stream, qf, qh, NH, totQ);
    hipLaunchKernelGGL(bn_rope, dim3(totK / 256), dim3(256), 0, stream, kf, kh, NKV, totK);
    hipLaunchKernelGGL(bn_vt,   dim3(totK / 256), dim3(256), 0, stream, vf, vth, totK);
  }

  {
    int blocks = BATCH * NH * (SEQ / 64);
    hipLaunchKernelGGL(bn_flash, dim3(blocks), dim3(128), 0, stream,
                       qh, kh, vth, mask, attn_out);
  }

  hipLaunchKernelGGL(bn_actq1, dim3(TOK), dim3(256), 0, stream, attn_out, no, ao8, aoinv);
  {
    int tiles = (TOK / 16) * (HDIM / 64);
    hipLaunchKernelGGL(bn_gemm_iu8, dim3(tiles / 8), dim3(256), 0, stream,
                       ao8, w8o, aoinv, acc + 6, 1.0f / nQO, out, TOK, HDIM, HDIM);
  }
}
